// NeuralVoxelHash_71691594104869
// MI455X (gfx1250) — matrix operand order repaired and checked by reading, not compiled, not run on hardware
//
#include <hip/hip_runtime.h>

typedef __attribute__((ext_vector_type(16))) _Float16 v16h;
typedef __attribute__((ext_vector_type(8)))  _Float16 v8h;
typedef __attribute__((ext_vector_type(4)))  _Float16 v4h;
typedef __attribute__((ext_vector_type(8)))  float    v8f;

#define NQ      1000000
#define NLEV    4
#define BUFSZ   4194304
#define BUFMASK 0x3FFFFFu   // BUFFER_SIZE = 2^22 -> int64 mod == 32-bit wrap & mask
#define NFEAT   500000
#define FDIM    8

typedef __attribute__((address_space(1))) int as1_int;
typedef __attribute__((address_space(3))) int as3_int;

union FragH { v16h v; v8h h8[2]; unsigned u[8]; _Float16 h[16]; };
union Pk4  { v4h v; unsigned u[2]; };

__device__ __forceinline__ unsigned pk2h(float a, float b) {
    return __builtin_bit_cast(unsigned, __builtin_amdgcn_cvt_pkrtz(a, b));
}

// Each wave handles 16 points; lane pair (r, r+16) shares point r
// (lane r gathers corners 0-3, lane r+16 corners 4-7). Per level, 4 WMMAs
// with block-diagonal A (packed weights, masked ANDs) x B (f16 features
// staged in LDS in B-fragment order: contiguous k = 16h + s, i.e. 8*(point%4) + corner) accumulate
// D[point, dim] across all levels.
__global__ __launch_bounds__(256)
void nvh_wmma_kernel(const float* __restrict__ qpts,
                     const float* __restrict__ feats,
                     const int*   __restrict__ fidx,
                     float*       __restrict__ out)
{
    __shared__ __align__(16) float    s_q[128 * 3];
    // dim axis padded 8->9: point stride = 36 dwords -> conflict-free scatter
    __shared__ __align__(16) _Float16 s_feat[8][16][9][8]; // [wave][pt][dim(+pad)][corner]

    const int tid  = threadIdx.x;
    const int wv   = tid >> 5;      // wave id (8 waves)
    const int lane = tid & 31;
    const int hw   = lane >> 4;     // half-wave
    const int r    = lane & 15;     // point row within wave tile / D column
    const int tile = blockIdx.x * 128;

    // ---- stage 128 query points (384 floats) into LDS (async on CDNA5) ----
#if __has_builtin(__builtin_amdgcn_global_load_async_to_lds_b32)
    for (int i = tid; i < 384; i += 256) {
        int gi = tile * 3 + i;
        if (gi < NQ * 3) {
            __builtin_amdgcn_global_load_async_to_lds_b32(
                (as1_int*)(qpts + gi), (as3_int*)(&s_q[i]), 0, 0);
        } else {
            s_q[i] = 0.0f;   // keep OOB lanes finite (NaN would poison WMMA)
        }
    }
#if __has_builtin(__builtin_amdgcn_s_wait_asynccnt)
    __builtin_amdgcn_s_wait_asynccnt(0);
#else
    asm volatile("s_wait_asynccnt 0" ::: "memory");
#endif
#else
    for (int i = tid; i < 384; i += 256) {
        int gi = tile * 3 + i;
        s_q[i] = (gi < NQ * 3) ? qpts[gi] : 0.0f;
    }
#endif
    __syncthreads();

    const int   ptw = wv * 16 + r;
    const int   ptg = tile + ptw;
    const float qx = s_q[ptw * 3 + 0];
    const float qy = s_q[ptw * 3 + 1];
    const float qz = s_q[ptw * 3 + 2];

    // ---- hoist ALL hash-index gathers (longest latency chain) ----
    int   ids[NLEV][4];
    float tfr[NLEV][3];
#pragma unroll
    for (int lev = 0; lev < NLEV; ++lev) {
        const float vs = 0.1f * (float)(1 << lev);
        const float fx = qx / vs, fy = qy / vs, fz = qz / vs;
        const float gx = floorf(fx), gy = floorf(fy), gz = floorf(fz);
        tfr[lev][0] = fx - gx; tfr[lev][1] = fy - gy; tfr[lev][2] = fz - gz;
        const int ix = (int)gx, iy = (int)gy, iz = (int)gz;
        const int* ftab = fidx + lev * BUFSZ;
#pragma unroll
        for (int k = 0; k < 4; ++k) {
            const int c = hw * 4 + k;                 // STEPS bit2=x bit1=y bit0=z
            const unsigned cx = (unsigned)(ix + ((c >> 2) & 1));
            const unsigned cy = (unsigned)(iy + ((c >> 1) & 1));
            const unsigned cz = (unsigned)(iz + (c & 1));
            const unsigned key =
                (cx * 73856093u + cy * 19349669u + cz * 83492791u) & BUFMASK;
            ids[lev][k] = ftab[key];
        }
    }

    // loop-invariant block-diagonal A masks: row r active in K-block j of WMMA g
    unsigned am0[4], am1[4];
#pragma unroll
    for (int g = 0; g < 4; ++g) {
        am0[g] = (r == 4 * g + 2 * hw)     ? 0xFFFFFFFFu : 0u;
        am1[g] = (r == 4 * g + 2 * hw + 1) ? 0xFFFFFFFFu : 0u;
    }

    v8f acc;
#pragma unroll
    for (int t = 0; t < 8; ++t) acc[t] = 0.0f;
    float valid0 = 0.0f;
    const int dcol = r & 7;

#pragma unroll
    for (int lev = 0; lev < NLEV; ++lev) {
        const float* fbase = feats + (size_t)lev * NFEAT * FDIM;

        // gather this lane's 4 feature rows
        float fv[4][8];
        int mn = 0x7fffffff;
#pragma unroll
        for (int k = 0; k < 4; ++k) {
            int id = ids[lev][k];
            mn = id < mn ? id : mn;
            id = id < 0 ? 0 : id;                     // safe_idx
            const float* fr = fbase + (size_t)id * FDIM;
            const float4 lo = *(const float4*)(fr);
            const float4 hi = *(const float4*)(fr + 4);
            fv[k][0] = lo.x; fv[k][1] = lo.y; fv[k][2] = lo.z; fv[k][3] = lo.w;
            fv[k][4] = hi.x; fv[k][5] = hi.y; fv[k][6] = hi.z; fv[k][7] = hi.w;
        }
        // combine empty-mask across lane pair: ds_swizzle SWAPX16 (xor 0x10)
        const int  mn2 = __builtin_amdgcn_ds_swizzle(mn, 0x401f);
        const bool ok  = ((mn < mn2 ? mn : mn2) >= 0);
        if (lev == 0 && hw == 0) valid0 = ok ? 1.0f : 0.0f;

        // scatter features to LDS in B-fragment order (pk-converted, no conflicts)
        asm volatile("" ::: "memory");
#pragma unroll
        for (int d = 0; d < 8; ++d) {
            Pk4 h;
            h.u[0] = pk2h(fv[0][d], fv[1][d]);
            h.u[1] = pk2h(fv[2][d], fv[3][d]);
            *(v4h*)(&s_feat[wv][r][d][hw * 4]) = h.v;
        }
        asm volatile("s_wait_dscnt 0" ::: "memory");   // same-wave LDS RAW

        // trilinear weights (zeroed when level is masked -> jnp.where)
        const float tx = tfr[lev][0], ty = tfr[lev][1], tz = tfr[lev][2];
        const float m  = ok ? 1.0f : 0.0f;
        const float x0 = (1.0f - tx) * m, x1 = tx * m;
        const float y0 = 1.0f - ty, y1 = ty, z0 = 1.0f - tz, z1 = tz;
        unsigned wp[4];
        wp[0] = pk2h(x0 * y0 * z0, x0 * y0 * z1);
        wp[1] = pk2h(x0 * y1 * z0, x0 * y1 * z1);
        wp[2] = pk2h(x1 * y0 * z0, x1 * y0 * z1);
        wp[3] = pk2h(x1 * y1 * z0, x1 * y1 * z1);

        // 4 WMMAs cover K = 16 points x 8 corners (block-diagonal A via ANDs).
        // A/B double-buffered on g&1 so fragment builds never overwrite the
        // registers an in-flight WMMA is reading (kills the 4x v_nop hazard).
        FragH Af[2], Bf[2];
#pragma unroll
        for (int g = 0; g < 4; ++g) {
            FragH& A = Af[g & 1];
            FragH& B = Bf[g & 1];
            const int p0 = 4 * g + 2 * hw;
            // B: lane column = r&7 (cols 8-15 duplicate 0-7; never stored)
            B.h8[0] = *(const v8h*)(&s_feat[wv][p0    ][dcol][0]);
            B.h8[1] = *(const v8h*)(&s_feat[wv][p0 + 1][dcol][0]);
#pragma unroll
            for (int t = 0; t < 4; ++t) {
                A.u[t]     = wp[t] & am0[g];
                A.u[4 + t] = wp[t] & am1[g];
            }
            acc = __builtin_amdgcn_wmma_f32_16x16x32_f16(
                false, A.v, false, B.v, (short)0, acc, false, false);
        }
    }

    // D layout: VGPR v -> row v + 8*hw, lane%16 -> column
    if (r < 8) {
#pragma unroll
        for (int v = 0; v < 8; ++v) {
            const int p = tile + wv * 16 + v + 8 * hw;
            if (p < NQ) out[(size_t)p * 8 + r] = acc[v];
        }
    }
    if (hw == 0 && ptg < NQ) out[(size_t)NQ * 8 + ptg] = valid0;
}

extern "C" void kernel_launch(void* const* d_in, const int* in_sizes, int n_in,
                              void* d_out, int out_size, void* d_ws, size_t ws_size,
                              hipStream_t stream)
{
    (void)in_sizes; (void)n_in; (void)d_ws; (void)ws_size; (void)out_size;
    const float* q  = (const float*)d_in[0];   // query_points (N,3) f32
    const float* f  = (const float*)d_in[1];   // features (4,500000,8) f32
    const int*   fi = (const int*)d_in[2];     // feature_indexs (4,4194304) int
    float*       o  = (float*)d_out;           // [N*8 sum_features | N valid]

    dim3 grid((NQ + 127) / 128);
    nvh_wmma_kernel<<<grid, dim3(256), 0, stream>>>(q, f, fi, o);
}
